// RNNModel_43911745634375
// MI455X (gfx1250) — compile-verified, matrix operand order checked by reading, not run on hardware
//
#include <hip/hip_runtime.h>
#include <hip/hip_bf16.h>
#include <math.h>

#define NTOKEN 32000
#define NINP   512
#define NHID   1024
#define SEQ    128
#define BATCH  64
#define ROWS   (SEQ * BATCH)        // 8192
#define G4H    (4 * NHID)           // 4096

typedef __attribute__((ext_vector_type(8)))  __bf16 v8bf;
typedef __attribute__((ext_vector_type(16))) __bf16 v16bf;
typedef __attribute__((ext_vector_type(8)))  float  v8f;
typedef __attribute__((ext_vector_type(4)))  unsigned int u32x4;
typedef __attribute__((ext_vector_type(8)))  int i32x8;
typedef __attribute__((ext_vector_type(4)))  int i32x4;

#define WMMA_BF16(a, b, c) \
    __builtin_amdgcn_wmma_f32_16x16x32_bf16(false, (a), false, (b), (short)0, (c), false, false)

// ---------------------------------------------------------------------------
// Fragment load: 16x32 bf16 tile (A or B^T), row-major, per ISA 16-bit layout:
//   lanes 0-15  : row = lane,     VGPR0-3 -> K=0..7,  VGPR4-7 -> K=16..23
//   lanes 16-31 : row = lane-16,  VGPR0-3 -> K=8..15, VGPR4-7 -> K=24..31
// => two contiguous 16-byte loads per lane (b128 global or ds).
// ---------------------------------------------------------------------------
__device__ __forceinline__ v16bf load_frag16x32(const __bf16* __restrict__ base,
                                                int ld, int rowbase, int kk, int lane) {
    int r     = rowbase + (lane & 15);
    int khalf = (lane >> 4) * 8;
    const __bf16* p = base + (size_t)r * ld + kk + khalf;
    v8bf lo = *(const v8bf*)(p);        // K = khalf + 0..7
    v8bf hi = *(const v8bf*)(p + 16);   // K = khalf + 16..23
    v16bf f;
#pragma unroll
    for (int i = 0; i < 8; ++i) { f[i] = lo[i]; f[i + 8] = hi[i]; }
    return f;
}

// ---------------------------------------------------------------------------
// TDM: issue a 2D tensor tile load (tile_rows x tile_k bf16, row stride =
// row_stride elements) from global into LDS at lds byte offset lds_off.
// D# packing per CDNA5 ISA 8.3/8.4: group0 = {count=1, lds_addr, global_addr,
// type=2}; group1 = {data_size=2B, tensor dims, tile dims, dim0 stride}.
// (6-arg builtin form: g0, g1, g2, g3, g4, cpol)
// ---------------------------------------------------------------------------
__device__ __forceinline__ void tdm_load_2d(unsigned lds_off, const __bf16* gptr,
                                            unsigned tile_k, unsigned tile_rows,
                                            unsigned row_stride) {
    unsigned long long ga = (unsigned long long)(uintptr_t)gptr;
    u32x4 g0;
    g0[0] = 1u;                                                // count=1, user desc
    g0[1] = lds_off;                                           // lds_addr (bytes)
    g0[2] = (unsigned)(ga & 0xFFFFFFFFu);                      // global_addr lo
    g0[3] = (unsigned)((ga >> 32) & 0x01FFFFFFu) | (2u << 30); // global_addr hi | type=2
    i32x8 g1;
    g1[0] = (int)(1u << 16);                                   // data_size = 1 -> 2 bytes
    g1[1] = (int)((row_stride & 0xFFFFu) << 16);               // tensor_dim0 lo
    g1[2] = (int)((row_stride >> 16) | ((tile_rows & 0xFFFFu) << 16)); // dim0 hi | tensor_dim1 lo
    g1[3] = (int)((tile_rows >> 16) | (tile_k << 16));         // dim1 hi | tile_dim0
    g1[4] = (int)(tile_rows & 0xFFFFu);                        // tile_dim1 (tile_dim2=0)
    g1[5] = (int)row_stride;                                   // tensor_dim0_stride lo32
    g1[6] = 0;                                                 // stride0 hi | stride1 lo
    g1[7] = 0;
    i32x4 z4 = {0, 0, 0, 0};
    i32x8 z8 = {0, 0, 0, 0, 0, 0, 0, 0};
    __builtin_amdgcn_tensor_load_to_lds(g0, g1, z4, z4, z8, 0);
}

// ---------------------------------------------------------------------------
// TDM + LDS double-buffered bf16 WMMA GEMM:
//   C[M,N] = A[M,K] * Bt[N,K]^T (+bias[n]) (+addend[m,n])
// Block = 256 threads (8 waves). Block tile = 64 rows x 128 cols, K-chunks of
// 64. Wave computes 32x32 (2x2 WMMA tiles). Wave 0 drives the TDM pipeline:
// chunk c+1 DMAs while all waves compute chunk c from LDS.
// grid = (N/128, M/64)
// ---------------------------------------------------------------------------
__global__ __launch_bounds__(256)
void gemm_bf16_wmma_tdm(const __bf16* __restrict__ A, const __bf16* __restrict__ Bt,
                        float* __restrict__ C,
                        const float* __restrict__ addend, const float* __restrict__ bias,
                        int K, int lda, int ldb, int ldc) {
    __shared__ __bf16 sA[2][64 * 64];    // 16 KB
    __shared__ __bf16 sB[2][128 * 64];   // 32 KB
    const int wave  = threadIdx.x >> 5;
    const int lane  = threadIdx.x & 31;
    const int mbase = blockIdx.y * 64;
    const int nbase = blockIdx.x * 128;
    const int mw    = (wave & 1) * 32;   // wave row offset in block tile
    const int nw    = (wave >> 1) * 32;  // wave col offset in block tile

    v8f acc00 = {}, acc01 = {}, acc10 = {}, acc11 = {};

    const int nchunk = K / 64;
    if (wave == 0) {
        tdm_load_2d((unsigned)(uintptr_t)&sA[0][0], A  + (size_t)mbase * lda, 64, 64,  (unsigned)lda);
        tdm_load_2d((unsigned)(uintptr_t)&sB[0][0], Bt + (size_t)nbase * ldb, 64, 128, (unsigned)ldb);
    }
    for (int c = 0; c < nchunk; ++c) {
        const int cur = c & 1;
        if (wave == 0) {
            if (c + 1 < nchunk) {
                const int nxt = cur ^ 1;
                tdm_load_2d((unsigned)(uintptr_t)&sA[nxt][0],
                            A  + (size_t)mbase * lda + (size_t)(c + 1) * 64, 64, 64,  (unsigned)lda);
                tdm_load_2d((unsigned)(uintptr_t)&sB[nxt][0],
                            Bt + (size_t)nbase * ldb + (size_t)(c + 1) * 64, 64, 128, (unsigned)ldb);
                __builtin_amdgcn_s_wait_tensorcnt(2);   // current chunk's pair done
            } else {
                __builtin_amdgcn_s_wait_tensorcnt(0);
            }
        }
        __syncthreads();                                 // chunk c resident in LDS
        const __bf16* tA = &sA[cur][0];
        const __bf16* tB = &sB[cur][0];
#pragma unroll
        for (int ks = 0; ks < 64; ks += 32) {
            v16bf a0 = load_frag16x32(tA, 64, mw,      ks, lane);
            v16bf a1 = load_frag16x32(tA, 64, mw + 16, ks, lane);
            v16bf b0 = load_frag16x32(tB, 64, nw,      ks, lane);
            v16bf b1 = load_frag16x32(tB, 64, nw + 16, ks, lane);
            acc00 = WMMA_BF16(a0, b0, acc00);
            acc01 = WMMA_BF16(a0, b1, acc01);
            acc10 = WMMA_BF16(a1, b0, acc10);
            acc11 = WMMA_BF16(a1, b1, acc11);
        }
        __syncthreads();                                 // done with buffer before reuse
    }

    // C/D layout: vgpr r, lanes 0-15: M=r, N=lane; lanes 16-31: M=8+r, N=lane-16
    const int mrow  = mbase + mw + (lane >> 4) * 8;
    const int ncol0 = nbase + nw + (lane & 15);
    v8f accs[2][2] = {{acc00, acc01}, {acc10, acc11}};
#pragma unroll
    for (int i = 0; i < 2; ++i) {
#pragma unroll
        for (int j = 0; j < 2; ++j) {
            int n = ncol0 + j * 16;
            float bv = bias ? bias[n] : 0.0f;
#pragma unroll
            for (int r = 0; r < 8; ++r) {
                size_t idx = (size_t)(mrow + i * 16 + r) * ldc + n;
                float v = accs[i][j][r] + bv;
                if (addend) v += addend[idx];
                C[idx] = v;
            }
        }
    }
}

// ---------------------------------------------------------------------------
// Direct-global bf16 WMMA GEMM (no LDS): used for the latency-critical
// recurrent step (M=64; the 8 MB weight matrix is L2-resident across steps).
// Block = 8 waves; wave computes 16x64; block tile 32x256. grid=(N/256, M/32)
// ---------------------------------------------------------------------------
__global__ __launch_bounds__(256)
void gemm_bf16_wmma(const __bf16* __restrict__ A, const __bf16* __restrict__ Bt,
                    float* __restrict__ C,
                    const float* __restrict__ addend, const float* __restrict__ bias,
                    int K, int lda, int ldb, int ldc) {
    const int wave = threadIdx.x >> 5;
    const int lane = threadIdx.x & 31;
    const int mblk = blockIdx.y * 32 + (wave & 1) * 16;
    const int nblk = blockIdx.x * 256 + (wave >> 1) * 64;

    v8f acc0 = {}, acc1 = {}, acc2 = {}, acc3 = {};

    for (int kk = 0; kk < K; kk += 32) {
        v16bf a  = load_frag16x32(A,  lda, mblk, kk, lane);
        v16bf b0 = load_frag16x32(Bt, ldb, nblk +  0, kk, lane);
        v16bf b1 = load_frag16x32(Bt, ldb, nblk + 16, kk, lane);
        v16bf b2 = load_frag16x32(Bt, ldb, nblk + 32, kk, lane);
        v16bf b3 = load_frag16x32(Bt, ldb, nblk + 48, kk, lane);
        acc0 = WMMA_BF16(a, b0, acc0);
        acc1 = WMMA_BF16(a, b1, acc1);
        acc2 = WMMA_BF16(a, b2, acc2);
        acc3 = WMMA_BF16(a, b3, acc3);
    }

    const int mrow  = mblk + (lane >> 4) * 8;
    const int ncol0 = lane & 15;
    v8f accs[4] = {acc0, acc1, acc2, acc3};
#pragma unroll
    for (int t = 0; t < 4; ++t) {
        int n = nblk + t * 16 + ncol0;
        float bv = bias ? bias[n] : 0.0f;
#pragma unroll
        for (int r = 0; r < 8; ++r) {
            size_t idx = (size_t)(mrow + r) * ldc + n;
            float v = accs[t][r] + bv;
            if (addend) v += addend[idx];
            C[idx] = v;
        }
    }
}

// ---------------------------------------------------------------------------
// Weight prep kernels
// ---------------------------------------------------------------------------
__global__ void transpose_to_bf16(const float* __restrict__ in, __bf16* __restrict__ out,
                                  int K, int N) {  // in: KxN row-major -> out: NxK
    size_t idx = (size_t)blockIdx.x * blockDim.x + threadIdx.x;
    if (idx >= (size_t)K * N) return;
    int n = (int)(idx % N);
    int k = (int)(idx / N);
    out[(size_t)n * K + k] = (__bf16)in[idx];
}

__global__ void convert_to_bf16(const float* __restrict__ in, __bf16* __restrict__ out,
                                size_t n) {
    size_t idx = (size_t)blockIdx.x * blockDim.x + threadIdx.x;
    if (idx < n) out[idx] = (__bf16)in[idx];
}

// Embedding gather + bf16 convert: out[(s*BATCH+b)*NINP + i]
__global__ void embed_bf16(const int* __restrict__ tokens, const float* __restrict__ emb_w,
                           __bf16* __restrict__ out) {
    size_t idx = (size_t)blockIdx.x * blockDim.x + threadIdx.x;  // ROWS*NINP threads
    int row = (int)(idx >> 9);          // NINP = 512
    int col = (int)(idx & 511);
    int tok = tokens[row];
    out[idx] = (__bf16)emb_w[(size_t)tok * NINP + col];
}

// ---------------------------------------------------------------------------
// LSTM state init + pointwise step
// ---------------------------------------------------------------------------
__global__ void lstm_init(const float* __restrict__ h0, const float* __restrict__ c0,
                          float* __restrict__ h_f32, float* __restrict__ c_f32,
                          __bf16* __restrict__ h_bf16) {
    int idx = blockIdx.x * blockDim.x + threadIdx.x;  // BATCH*NHID
    float h = h0[idx];
    h_f32[idx] = h;
    c_f32[idx] = c0[idx];
    h_bf16[idx] = (__bf16)h;
}

__global__ void lstm_pointwise(const float* __restrict__ gates, float* __restrict__ c_state,
                               float* __restrict__ h_f32, __bf16* __restrict__ h_bf16,
                               __bf16* __restrict__ hseq_t) {
    int idx = blockIdx.x * blockDim.x + threadIdx.x;  // BATCH*NHID
    int b = idx >> 10;                                 // NHID = 1024
    int j = idx & 1023;
    const float* g = gates + (size_t)b * G4H;
    float ig = 1.0f / (1.0f + __expf(-g[j]));
    float fg = 1.0f / (1.0f + __expf(-g[NHID + j]));
    float gg = tanhf(g[2 * NHID + j]);
    float og = 1.0f / (1.0f + __expf(-g[3 * NHID + j]));
    float c  = fg * c_state[idx] + ig * gg;
    float h  = og * tanhf(c);
    c_state[idx] = c;
    h_f32[idx]   = h;
    __bf16 hb = (__bf16)h;
    h_bf16[idx] = hb;
    hseq_t[idx] = hb;
}

__global__ void copy_hc_out(const float* __restrict__ h, const float* __restrict__ c,
                            float* __restrict__ out) {
    int idx = blockIdx.x * blockDim.x + threadIdx.x;  // 2*BATCH*NHID
    const int HC = BATCH * NHID;
    out[idx] = (idx < HC) ? h[idx] : c[idx - HC];
}

// ---------------------------------------------------------------------------
// Launch
// ---------------------------------------------------------------------------
extern "C" void kernel_launch(void* const* d_in, const int* in_sizes, int n_in,
                              void* d_out, int out_size, void* d_ws, size_t ws_size,
                              hipStream_t stream) {
    const int*   tokens   = (const int*)  d_in[0];  // (SEQ, BATCH)
    const float* h0       = (const float*)d_in[1];  // (BATCH, NHID)
    const float* c0       = (const float*)d_in[2];
    const float* emb_w    = (const float*)d_in[3];  // (NTOKEN, NINP)
    const float* w_ih     = (const float*)d_in[4];  // (NINP, 4H)
    const float* w_hh     = (const float*)d_in[5];  // (NHID, 4H)
    const float* bias     = (const float*)d_in[6];  // (4H)
    const float* dec_w    = (const float*)d_in[7];  // (NTOKEN, NHID)  already NxK
    const float* dec_b    = (const float*)d_in[8];  // (NTOKEN)

    float* logits = (float*)d_out;                              // (SEQ,BATCH,NTOKEN)
    float* hc_out = (float*)d_out + (size_t)ROWS * NTOKEN;      // h_t then c_t

    // ---- workspace carve-up (all offsets 256B aligned) ----
    char* ws = (char*)d_ws;
    size_t off = 0;
    __bf16* w_ih_t  = (__bf16*)(ws + off); off += (size_t)G4H * NINP * 2;        // (4H, NINP)
    __bf16* w_hh_t  = (__bf16*)(ws + off); off += (size_t)G4H * NHID * 2;        // (4H, NHID)
    __bf16* dec_wb  = (__bf16*)(ws + off); off += (size_t)NTOKEN * NHID * 2;     // (V, NHID)
    __bf16* emb_b   = (__bf16*)(ws + off); off += (size_t)ROWS * NINP * 2;       // (ROWS, NINP)
    float*  x_proj  = (float*) (ws + off); off += (size_t)ROWS * G4H * 4;        // (ROWS, 4H)
    __bf16* hseq    = (__bf16*)(ws + off); off += (size_t)ROWS * NHID * 2;       // (ROWS, NHID)
    float*  gates   = (float*) (ws + off); off += (size_t)BATCH * G4H * 4;
    __bf16* h_bf    = (__bf16*)(ws + off); off += (size_t)BATCH * NHID * 2;
    float*  h_f     = (float*) (ws + off); off += (size_t)BATCH * NHID * 4;
    float*  c_f     = (float*) (ws + off); off += (size_t)BATCH * NHID * 4;

    // ---- weight conversion ----
    {
        size_t n = (size_t)NINP * G4H;
        transpose_to_bf16<<<dim3((n + 255) / 256), dim3(256), 0, stream>>>(w_ih, w_ih_t, NINP, G4H);
    }
    {
        size_t n = (size_t)NHID * G4H;
        transpose_to_bf16<<<dim3((n + 255) / 256), dim3(256), 0, stream>>>(w_hh, w_hh_t, NHID, G4H);
    }
    {
        size_t n = (size_t)NTOKEN * NHID;
        convert_to_bf16<<<dim3((n + 255) / 256), dim3(256), 0, stream>>>(dec_w, dec_wb, n);
    }

    // ---- embedding gather ----
    {
        size_t n = (size_t)ROWS * NINP;
        embed_bf16<<<dim3((n + 255) / 256), dim3(256), 0, stream>>>(tokens, emb_w, emb_b);
    }

    // ---- x_proj = emb @ W_ih + bias : M=8192, N=4096, K=512 (TDM pipeline) ----
    gemm_bf16_wmma_tdm<<<dim3(G4H / 128, ROWS / 64), dim3(256), 0, stream>>>(
        emb_b, w_ih_t, x_proj, nullptr, bias, NINP, NINP, NINP, G4H);

    // ---- LSTM state init ----
    lstm_init<<<dim3(BATCH * NHID / 256), dim3(256), 0, stream>>>(h0, c0, h_f, c_f, h_bf);

    // ---- 128 recurrent steps (latency-bound: direct-global WMMA, L2-resident W_hh) ----
    for (int t = 0; t < SEQ; ++t) {
        const float* xp_t = x_proj + (size_t)t * BATCH * G4H;
        // gates = h @ W_hh + x_proj[t] : M=64, N=4096, K=1024
        gemm_bf16_wmma<<<dim3(G4H / 256, BATCH / 32), dim3(256), 0, stream>>>(
            h_bf, w_hh_t, gates, xp_t, nullptr, NHID, NHID, NHID, G4H);
        lstm_pointwise<<<dim3(BATCH * NHID / 256), dim3(256), 0, stream>>>(
            gates, c_f, h_f, h_bf, hseq + (size_t)t * BATCH * NHID);
    }

    // ---- decoder: logits = hidden @ dec_w^T + dec_b : M=8192, N=32000, K=1024 ----
    gemm_bf16_wmma_tdm<<<dim3(NTOKEN / 128, ROWS / 64), dim3(256), 0, stream>>>(
        hseq, dec_wb, logits, nullptr, dec_b, NHID, NHID, NHID, NTOKEN);

    // ---- final state outputs ----
    copy_hc_out<<<dim3(2 * BATCH * NHID / 256), dim3(256), 0, stream>>>(h_f, c_f, hc_out);
}
